// GATEncoder_50208167690742
// MI455X (gfx1250) — matrix operand order repaired and checked by reading, not compiled, not run on hardware
//
#include <hip/hip_runtime.h>

// ---------------------------------------------------------------------------
// GAT encoder (2 layers + MLP) for MI455X / gfx1250, wave32 + WMMA.
// Dense GEMMs -> v_wmma_f32_16x16x32_bf16 ; head projection -> v_wmma_f32_16x16x4_f32.
// Segment softmax / aggregation -> L2 float atomics.
// ---------------------------------------------------------------------------

typedef __attribute__((ext_vector_type(16))) __bf16 v16bf;
typedef __attribute__((ext_vector_type(8)))  __bf16 v8bf;
typedef __attribute__((ext_vector_type(8)))  float  v8f;
typedef __attribute__((ext_vector_type(2)))  float  v2f;

#define NEG_SLOPE 0.1f

#if defined(__has_builtin)
#if __has_builtin(__builtin_amdgcn_wmma_f32_16x16x4_f32)
#define HAVE_WMMA_F32X4 1
#endif
#endif

// ---- order-preserving float<->uint map for atomic segment-max -------------
__device__ __forceinline__ unsigned fenc(float f) {
  unsigned u = __float_as_uint(f);
  return (u & 0x80000000u) ? ~u : (u | 0x80000000u);
}
__device__ __forceinline__ float fdec(unsigned u) {
  return (u & 0x80000000u) ? __uint_as_float(u & 0x7fffffffu)
                           : __uint_as_float(~u);
}

// ---- fp32 -> bf16 with arbitrary destination stride/offset ----------------
__global__ void k_f32_to_bf16(const float* __restrict__ src, __bf16* __restrict__ dst,
                              int rows, int cols, int dstStride, int dstOff) {
  int i = blockIdx.x * blockDim.x + threadIdx.x;
  if (i >= rows * cols) return;
  int r = i / cols, c = i - r * cols;
  dst[(size_t)r * dstStride + dstOff + c] = (__bf16)src[i];
}

// ---- C[M,NC] = A[M,K](bf16) @ W[NC,K]^T(bf16) + bias, fp32 accum ----------
// One wave computes a 16x64 strip: 4 col-tiles reuse one A fragment.
__global__ void k_gemm_bf16(const __bf16* __restrict__ A, const __bf16* __restrict__ W,
                            const float* __restrict__ bias, float* __restrict__ C,
                            int M, int K, int NC) {
  int wid  = (blockIdx.x * blockDim.x + threadIdx.x) >> 5;
  int lane = threadIdx.x & 31;
  int colGroups = NC >> 6;          // groups of 64 columns
  int mTiles    = M >> 4;
  if (wid >= mTiles * colGroups) return;
  int mt = wid / colGroups, jg = wid - mt * colGroups;
  int m0 = mt << 4, j0 = jg << 6;
  int lm = lane & 15, hl = lane >> 4;

  const __bf16* arow = A + (size_t)(m0 + lm) * K;
  v8f acc[4] = {};

  for (int k0 = 0; k0 < K; k0 += 32) {
    // A fragment: 16x32, row lm; contiguous 16 bf16 at k0+16*hl
    v16bf a = *(const v16bf*)(arow + k0 + hl * 16);
#pragma unroll
    for (int jt = 0; jt < 4; ++jt) {
      // B fragment: col (j0+jt*16+lm) = W row; contiguous 16 bf16 at k0+16*hl
      v16bf b = *(const v16bf*)(W + (size_t)(j0 + jt * 16 + lm) * K + k0 + hl * 16);
      acc[jt] = __builtin_amdgcn_wmma_f32_16x16x32_bf16(
          false, a, false, b, (short)0, acc[jt], false, false);
    }
  }
#pragma unroll
  for (int jt = 0; jt < 4; ++jt) {
    int col = j0 + jt * 16 + lm;
    float bv = bias[col];
#pragma unroll
    for (int r = 0; r < 8; ++r)
      C[(size_t)(m0 + r + hl * 8) * NC + col] = acc[jt][r] + bv;
  }
}

// ---- attention logits: el/er[n,h] = sum_d feat[n,h,d] * al/ar[h,d] --------
__global__ void k_elr(const float* __restrict__ feat, const float* __restrict__ al,
                      const float* __restrict__ ar, float* __restrict__ el,
                      float* __restrict__ er, int N) {
  int t = blockIdx.x * blockDim.x + threadIdx.x;
  if (t >= N * 8) return;
  int n = t >> 3, h = t & 7;
  const float* f = feat + (size_t)n * 256 + h * 32;
  const float* A = al + h * 32;
  const float* B = ar + h * 32;
  float sl = 0.f, sr = 0.f;
#pragma unroll
  for (int i = 0; i < 32; ++i) {
    float fv = f[i];
    sl = fmaf(fv, A[i], sl);
    sr = fmaf(fv, B[i], sr);
  }
  el[t] = sl;
  er[t] = sr;
}

// ---- edge score + segment max ---------------------------------------------
__global__ void k_edge_score(const float* __restrict__ el, const float* __restrict__ er,
                             const int* __restrict__ src, const int* __restrict__ dst,
                             const float* __restrict__ w, float* __restrict__ esc,
                             unsigned* __restrict__ mx, int E) {
  int i = blockIdx.x * blockDim.x + threadIdx.x;
  if (i >= E * 8) return;
  int e = i >> 3, h = i & 7;
  int s = src[e], t = dst[e];
  float v = el[s * 8 + h] + er[t * 8 + h];
  v = v > 0.f ? v : NEG_SLOPE * v;
  v *= w[e];
  esc[i] = v;
  atomicMax(&mx[t * 8 + h], fenc(v));
}

// ---- exp(e - max) + segment sum (in-place: esc becomes alpha-numerator) ---
__global__ void k_edge_exp(float* __restrict__ esc, const unsigned* __restrict__ mx,
                           const int* __restrict__ dst, float* __restrict__ denom, int E) {
  int i = blockIdx.x * blockDim.x + threadIdx.x;
  if (i >= E * 8) return;
  int e = i >> 3, h = i & 7;
  int t = dst[e];
  float av = __expf(esc[i] - fdec(mx[t * 8 + h]));
  esc[i] = av;
  atomicAdd(&denom[t * 8 + h], av);
}

// ---- scatter: agg[dst] += feat[src] * alpha  (one wave per edge) ----------
__global__ void k_scatter(const float* __restrict__ feat, const float* __restrict__ a,
                          const float* __restrict__ denom, const int* __restrict__ src,
                          const int* __restrict__ dst, float* __restrict__ agg, int E) {
  int wave = (blockIdx.x * blockDim.x + threadIdx.x) >> 5;
  int lane = threadIdx.x & 31;
  if (wave >= E) return;
  int s = src[wave], t = dst[wave];
  const float* fs = feat + (size_t)s * 256;
  float* ag = agg + (size_t)t * 256;
#pragma unroll
  for (int h = 0; h < 8; ++h) {
    float coef = a[(size_t)wave * 8 + h] / denom[(size_t)t * 8 + h];
    atomicAdd(&ag[h * 32 + lane], fs[h * 32 + lane] * coef);
  }
}

// ---- head projection: out[Mrows,32] = agg[Mrows,32] @ Wout^T + bout, PReLU
// fp32 WMMA (16x16x4), K=32 -> 8 steps, 2 col tiles per wave.
__global__ void k_wout_prelu(const float* __restrict__ agg, const float* __restrict__ Wout,
                             const float* __restrict__ bout, const float* __restrict__ prelu,
                             int layer, float* __restrict__ out, int Mrows) {
  int wid  = (blockIdx.x * blockDim.x + threadIdx.x) >> 5;
  int lane = threadIdx.x & 31;
  if (wid >= (Mrows >> 4)) return;
  int m0 = wid << 4;
  int lm = lane & 15, hl = lane >> 4;
  const float* arow = agg + (size_t)(m0 + lm) * 32;
  v8f acc0 = {}, acc1 = {};
#if defined(HAVE_WMMA_F32X4)
#pragma unroll
  for (int k0 = 0; k0 < 32; k0 += 4) {
    v2f a  = *(const v2f*)(arow + k0 + hl * 2);
    v2f b0 = *(const v2f*)(Wout + (size_t)lm * 32 + k0 + hl * 2);
    v2f b1 = *(const v2f*)(Wout + (size_t)(16 + lm) * 32 + k0 + hl * 2);
    acc0 = __builtin_amdgcn_wmma_f32_16x16x4_f32(false, a, false, b0, (short)0, acc0, false, false);
    acc1 = __builtin_amdgcn_wmma_f32_16x16x4_f32(false, a, false, b1, (short)0, acc1, false, false);
  }
#else
#pragma unroll
  for (int r = 0; r < 8; ++r) {
    const float* rrow = agg + (size_t)(m0 + r + hl * 8) * 32;
    float s0 = 0.f, s1 = 0.f;
#pragma unroll
    for (int k = 0; k < 32; ++k) {
      s0 = fmaf(rrow[k], Wout[lm * 32 + k], s0);
      s1 = fmaf(rrow[k], Wout[(16 + lm) * 32 + k], s1);
    }
    acc0[r] = s0;
    acc1[r] = s1;
  }
#endif
  float pa  = prelu[layer];
  float b0v = bout[lm], b1v = bout[16 + lm];
#pragma unroll
  for (int r = 0; r < 8; ++r) {
    int row = m0 + r + hl * 8;
    float v0 = acc0[r] + b0v;
    float v1 = acc1[r] + b1v;
    v0 = v0 >= 0.f ? v0 : pa * v0;
    v1 = v1 >= 0.f ? v1 : pa * v1;
    out[(size_t)row * 32 + lm]      = v0;
    out[(size_t)row * 32 + 16 + lm] = v1;
  }
}

// ---------------------------------------------------------------------------
extern "C" void kernel_launch(void* const* d_in, const int* in_sizes, int n_in,
                              void* d_out, int out_size, void* d_ws, size_t ws_size,
                              hipStream_t stream) {
  (void)n_in; (void)out_size; (void)ws_size;
  const float* features = (const float*)d_in[0];
  const int*   src      = (const int*)d_in[1];
  const int*   dstIdx   = (const int*)d_in[2];
  const float* w        = (const float*)d_in[3];
  const float* Wfc      = (const float*)d_in[4];
  const float* bfc      = (const float*)d_in[5];
  const float* al       = (const float*)d_in[6];
  const float* ar       = (const float*)d_in[7];
  const float* Wout     = (const float*)d_in[8];
  const float* bout     = (const float*)d_in[9];
  const float* prelu    = (const float*)d_in[10];
  const float* Wmlp     = (const float*)d_in[11];
  const float* bmlp     = (const float*)d_in[12];

  const int D = 256, H = 8;
  const int E = in_sizes[1];
  const int N = in_sizes[0] / D;   // B==1

  // ---- workspace carving (all buffers fully (re)written each call) -------
  char*  base = (char*)d_ws;
  size_t off  = 0;
  auto carve = [&](size_t bytes) -> void* {
    void* p = base + off;
    off += (bytes + 255) & ~(size_t)255;
    return p;
  };
  __bf16*   xb    = (__bf16*)carve((size_t)N * D * 2);       // layer input, bf16
  __bf16*   hcat  = (__bf16*)carve((size_t)N * 2 * D * 2);   // concat(out1,out2), bf16
  __bf16*   Wb    = (__bf16*)carve((size_t)D * 2 * D * 2);   // weight staging, bf16
  float*    feat  = (float*)carve((size_t)N * D * 4);
  float*    outl  = (float*)carve((size_t)N * D * 4);
  float*    agg   = (float*)carve((size_t)N * D * 4);
  float*    el    = (float*)carve((size_t)N * H * 4);
  float*    er    = (float*)carve((size_t)N * H * 4);
  unsigned* mx    = (unsigned*)carve((size_t)N * H * 4);
  float*    denom = (float*)carve((size_t)N * H * 4);
  float*    esc   = (float*)carve((size_t)E * H * 4);

  const int thr = 256;
  auto cdiv = [](long a, long b) { return (int)((a + b - 1) / b); };

  k_f32_to_bf16<<<cdiv((long)N * D, thr), thr, 0, stream>>>(features, xb, N, D, D, 0);

  for (int l = 0; l < 2; ++l) {
    // feat = x @ Wfc[l]^T + bfc[l]
    k_f32_to_bf16<<<cdiv((long)D * D, thr), thr, 0, stream>>>(Wfc + (size_t)l * D * D, Wb, D, D, D, 0);
    {
      long waves = (long)(N / 16) * (D / 64);
      k_gemm_bf16<<<cdiv(waves * 32, thr), thr, 0, stream>>>(xb, Wb, bfc + l * D, feat, N, D, D);
    }
    k_elr<<<cdiv((long)N * H, thr), thr, 0, stream>>>(feat, al + l * H * 32, ar + l * H * 32, el, er, N);

    hipMemsetAsync(mx,    0, (size_t)N * H * 4, stream);  // encoded 0 == -inf
    hipMemsetAsync(denom, 0, (size_t)N * H * 4, stream);
    hipMemsetAsync(agg,   0, (size_t)N * D * 4, stream);

    k_edge_score<<<cdiv((long)E * H, thr), thr, 0, stream>>>(el, er, src, dstIdx, w, esc, mx, E);
    k_edge_exp<<<cdiv((long)E * H, thr), thr, 0, stream>>>(esc, mx, dstIdx, denom, E);
    k_scatter<<<cdiv((long)E * 32, thr), thr, 0, stream>>>(feat, esc, denom, src, dstIdx, agg, E);

    // out = PReLU(agg @ Wout[l]^T + bout[l]), rows = N*H
    k_wout_prelu<<<cdiv((long)(N * H / 16) * 32, thr), thr, 0, stream>>>(
        agg, Wout + l * 32 * 32, bout + l * 32, prelu, l, outl, N * H);

    // stash into concat buffer (bf16) and feed next layer
    k_f32_to_bf16<<<cdiv((long)N * D, thr), thr, 0, stream>>>(outl, hcat, N, D, 2 * D, l * D);
    if (l == 0)
      k_f32_to_bf16<<<cdiv((long)N * D, thr), thr, 0, stream>>>(outl, xb, N, D, D, 0);
  }

  // final MLP: d_out = hcat @ Wmlp^T + bmlp   (K = 512)
  k_f32_to_bf16<<<cdiv((long)D * 2 * D, thr), thr, 0, stream>>>(Wmlp, Wb, D, 2 * D, 2 * D, 0);
  {
    long waves = (long)(N / 16) * (D / 64);
    k_gemm_bf16<<<cdiv(waves * 32, thr), thr, 0, stream>>>(hcat, Wb, bmlp, (float*)d_out, N, 2 * D, D);
  }
}
